// InrEncoderLayer_59107339927930
// MI455X (gfx1250) — matrix operand order repaired and checked by reading, not compiled, not run on hardware
//
#include <hip/hip_runtime.h>
#include <hip/hip_bf16.h>
#include <math.h>

typedef __bf16 bf16_t;
typedef __attribute__((ext_vector_type(16))) __bf16 v16bf;
typedef __attribute__((ext_vector_type(8)))  __bf16 v8bf;
typedef __attribute__((ext_vector_type(8)))  float  v8f;
typedef __attribute__((ext_vector_type(4)))  float  v4f;
typedef int v4i_gcc __attribute__((vector_size(16)));   // matches async-LDS builtin pointee

#define DEVINL __device__ __forceinline__

// CDNA5 async Global->LDS copy path (ASYNCcnt), guarded for toolchain support.
#if defined(__AMDGCN__) && __has_builtin(__builtin_amdgcn_global_load_async_to_lds_b128) && \
    __has_builtin(__builtin_amdgcn_s_wait_asynccnt)
#define USE_ASYNC_LDS 1
#else
#define USE_ASYNC_LDS 0
#endif

#if USE_ASYNC_LDS
DEVINL void async_copy16(const void* gsrc, void* ldst) {
  // generic->as(1): bit-identical; generic->as(3): low 32 bits are the LDS address
  __builtin_amdgcn_global_load_async_to_lds_b128(
      (__attribute__((address_space(1))) v4i_gcc*)(uintptr_t)gsrc,
      (__attribute__((address_space(3))) v4i_gcc*)(uint32_t)(uintptr_t)ldst,
      0, 0);
}
#endif

static constexpr int   B_   = 4;
static constexpr int   S_   = 1024;
static constexpr int   F_   = 1024;
static constexpr int   H_   = 16;
static constexpr int   Dh_  = 64;
static constexpr int   EXPF = 4096;       // EXP * F
static constexpr float NEGINF = -1e9f;
static constexpr float SCALE_INV = 0.03125f;   // 1/sqrt(F) = 1/32

// ---------------------------------------------------------------------------
// WMMA fragment helpers (bf16 16x16x32, wave32)
// A 16x32: lane = row (mod 16); lanes 0-15 hold K 0-7,16-23; lanes 16-31 hold K 8-15,24-31
// B 32x16: lane = col (mod 16); lanes 0-15 hold K 0-7,16-23; lanes 16-31 hold K 8-15,24-31
// C/D    : VGPR i holds row i (lanes 0-15) and row i+8 (lanes 16-31), col = lane&15
// ---------------------------------------------------------------------------
DEVINL v16bf frag_a(const bf16_t* lds, int ld, int row0, int kchunk, int lane) {
  const bf16_t* p = lds + (long)(row0 + (lane & 15)) * ld + kchunk * 32 + ((lane >> 4) * 8);
  v16bf a;
#pragma unroll
  for (int i = 0; i < 8; ++i) { a[i] = p[i]; a[8 + i] = p[16 + i]; }
  return a;
}

DEVINL v16bf frag_b(const bf16_t* lds, int ld, int col0, int kchunk, int lane) {
  const bf16_t* p = lds + (long)(kchunk * 32 + ((lane >> 4) * 8)) * ld + col0 + (lane & 15);
  v16bf b;
#pragma unroll
  for (int i = 0; i < 8; ++i) { b[i] = p[(long)i * ld]; b[8 + i] = p[(long)(16 + i) * ld]; }
  return b;
}

DEVINL v8f wmma_bf16(v16bf a, v16bf b, v8f c) {
  return __builtin_amdgcn_wmma_f32_16x16x32_bf16(false, a, false, b, (short)0, c,
                                                 false, false);
}

// ---------------------------------------------------------------------------
// fp32 -> bf16 elementwise convert
// ---------------------------------------------------------------------------
__global__ __launch_bounds__(256) void cvt_bf16_kernel(const float* __restrict__ in,
                                                       bf16_t* __restrict__ out,
                                                       long n) {
  long i = (long)blockIdx.x * 1024 + (long)threadIdx.x * 4;
  if (i + 3 < n) {
    v4f v = *(const v4f*)&in[i];
#pragma unroll
    for (int j = 0; j < 4; ++j) out[i + j] = (bf16_t)v[j];
  }
}

// ---------------------------------------------------------------------------
// LayerNorm over F=1024, one block (256 thr) per row, writes bf16
// ---------------------------------------------------------------------------
__global__ __launch_bounds__(256) void ln_kernel(const float* __restrict__ X,
                                                 const float* __restrict__ g,
                                                 const float* __restrict__ be,
                                                 bf16_t* __restrict__ out) {
  const long row = blockIdx.x;
  const int  tid = threadIdx.x, lane = tid & 31, wave = tid >> 5;
  const float* xr = X + row * (long)F_;
  v4f xv = *(const v4f*)&xr[tid * 4];
  float s  = xv[0] + xv[1] + xv[2] + xv[3];
  float s2 = xv[0] * xv[0] + xv[1] * xv[1] + xv[2] * xv[2] + xv[3] * xv[3];
#pragma unroll
  for (int off = 16; off >= 1; off >>= 1) {
    s  += __shfl_xor(s, off, 32);
    s2 += __shfl_xor(s2, off, 32);
  }
  __shared__ float red[16];
  if (lane == 0) { red[wave] = s; red[8 + wave] = s2; }
  __syncthreads();
  float ts = 0.f, ts2 = 0.f;
#pragma unroll
  for (int w = 0; w < 8; ++w) { ts += red[w]; ts2 += red[8 + w]; }
  const float mu   = ts * (1.0f / F_);
  const float var  = ts2 * (1.0f / F_) - mu * mu;
  const float rstd = rsqrtf(var + 1e-5f);
#pragma unroll
  for (int j = 0; j < 4; ++j) {
    int c = tid * 4 + j;
    out[row * F_ + c] = (bf16_t)((xv[j] - mu) * rstd * g[c] + be[c]);
  }
}

// ---------------------------------------------------------------------------
// r_mat masked row stats: rowmax and sum(exp(v-rowmax)), one block per (b,q)
// ---------------------------------------------------------------------------
__global__ __launch_bounds__(256) void rmstat_kernel(const float* __restrict__ r_mat,
                                                     const float* __restrict__ attn_mask,
                                                     const float* __restrict__ pad_mask,
                                                     float* __restrict__ rmax,
                                                     float* __restrict__ rsum) {
  const long row = blockIdx.x;               // b*S + q
  const long q   = row % S_;
  const int  tid = threadIdx.x, lane = tid & 31, wave = tid >> 5;
  const float* rr = r_mat    + row * (long)S_;
  const float* pr = pad_mask + row * (long)S_;
  const float* ar = attn_mask + q * (long)S_;
  float vals[4], mx = -3e38f;
#pragma unroll
  for (int j = 0; j < 4; ++j) {
    int t = tid * 4 + j;
    bool m = (pr[t] == 0.f) || (ar[t] == 0.f);
    vals[j] = m ? NEGINF : rr[t];
    mx = fmaxf(mx, vals[j]);
  }
#pragma unroll
  for (int off = 16; off >= 1; off >>= 1) mx = fmaxf(mx, __shfl_xor(mx, off, 32));
  __shared__ float smax[8], ssum[8];
  if (lane == 0) smax[wave] = mx;
  __syncthreads();
  float bmax = smax[0];
#pragma unroll
  for (int w = 1; w < 8; ++w) bmax = fmaxf(bmax, smax[w]);
  float sum = 0.f;
#pragma unroll
  for (int j = 0; j < 4; ++j) sum += __expf(vals[j] - bmax);
#pragma unroll
  for (int off = 16; off >= 1; off >>= 1) sum += __shfl_xor(sum, off, 32);
  if (lane == 0) ssum[wave] = sum;
  __syncthreads();
  if (tid == 0) {
    float t = 0.f;
#pragma unroll
    for (int w = 0; w < 8; ++w) t += ssum[w];
    rmax[row] = bmax;
    rsum[row] = t;
  }
}

// ---------------------------------------------------------------------------
// Generic batched bf16 WMMA GEMM, BK=32, 256 threads = 8 waves (2x4 wave grid)
// Double-buffered LDS staging; async Global->LDS (ASYNCcnt) when available.
// EPI: 0 = bf16 store; 1 = bf16 relu(C+bias); 2 = f32 Res+C; 3 = f32 Res+C+bias
// batch offsets: A += (z/adiv)*strideA ; B += (z%bmod)*strideB ; C/Res += z*stride
// ---------------------------------------------------------------------------
template <int BM, int BN, int EPI>
__global__ __launch_bounds__(256) void gemm_bf16_kernel(
    const bf16_t* __restrict__ A, long lda, long strideA, int adiv,
    const bf16_t* __restrict__ Bm, long ldb, long strideB, int bmod,
    void* __restrict__ Cout, long ldc, long strideC,
    const float* __restrict__ Res, long strideR,
    const float* __restrict__ bias, int K) {
  const int tid = threadIdx.x, lane = tid & 31, wave = tid >> 5;
  const int wr = wave >> 2, wc = wave & 3;
  const int z = blockIdx.z;
  A  += (long)(z / adiv) * strideA + (long)blockIdx.x * BM * lda;
  Bm += (long)(z % bmod) * strideB + (long)blockIdx.y * BN;

  constexpr int MI = BM / 32;         // wmma tiles per wave in M
  constexpr int NI = BN / 64;         // wmma tiles per wave in N
  constexpr int LDA_S = 48;           // 32 + 16 pad (bf16)
  constexpr int LDB_S = BN + 16;
  constexpr int BNC = BN / 8;         // 8-elem chunks per B row
  constexpr int ITA = (BM * 4) / 256; // per-thread 16B copies for A tile
  constexpr int ITB = (4 * BN) / 256; // per-thread 16B copies for B tile
  __shared__ alignas(128) bf16_t sA[2][BM * LDA_S];
  __shared__ alignas(128) bf16_t sB[2][32 * LDB_S];

  v8f acc[MI][NI];
#pragma unroll
  for (int mi = 0; mi < MI; ++mi)
#pragma unroll
    for (int ni = 0; ni < NI; ++ni) acc[mi][ni] = (v8f)0.f;

  auto stage = [&](int k0, int buf) {
#pragma unroll
    for (int it = 0; it < ITA; ++it) {
      int c = tid + it * 256;
      int row = c >> 2, seg = c & 3;
      const void* gp = &A[(long)row * lda + k0 + seg * 8];
      void* lp = &sA[buf][row * LDA_S + seg * 8];
#if USE_ASYNC_LDS
      async_copy16(gp, lp);
#else
      *(v8bf*)lp = *(const v8bf*)gp;
#endif
    }
#pragma unroll
    for (int it = 0; it < ITB; ++it) {
      int c = tid + it * 256;
      int row = c / BNC, seg = c % BNC;
      const void* gp = &Bm[(long)(k0 + row) * ldb + seg * 8];
      void* lp = &sB[buf][row * LDB_S + seg * 8];
#if USE_ASYNC_LDS
      async_copy16(gp, lp);
#else
      *(v8bf*)lp = *(const v8bf*)gp;
#endif
    }
  };

  auto compute = [&](int buf) {
    v16bf af[MI], bfv[NI];
#pragma unroll
    for (int mi = 0; mi < MI; ++mi)
      af[mi] = frag_a(&sA[buf][0], LDA_S, wr * (BM / 2) + mi * 16, 0, lane);
#pragma unroll
    for (int ni = 0; ni < NI; ++ni)
      bfv[ni] = frag_b(&sB[buf][0], LDB_S, wc * (BN / 4) + ni * 16, 0, lane);
#pragma unroll
    for (int mi = 0; mi < MI; ++mi)
#pragma unroll
      for (int ni = 0; ni < NI; ++ni)
        acc[mi][ni] = wmma_bf16(af[mi], bfv[ni], acc[mi][ni]);
  };

  // software pipeline: stage tile k+1 async while computing tile k
  stage(0, 0);
  int buf = 0;
  int k0 = 0;
  for (; k0 + 32 < K; k0 += 32) {
    stage(k0 + 32, buf ^ 1);
    if (k0 + 64 < K) {   // L2 prefetch two tiles ahead (global_prefetch_b8)
      __builtin_prefetch(&A[(long)(tid >> 1) * lda + k0 + 64], 0, 3);
      __builtin_prefetch(&Bm[(long)(k0 + 64 + (tid & 31)) * ldb], 0, 3);
    }
#if USE_ASYNC_LDS
    __builtin_amdgcn_s_wait_asynccnt(ITA + ITB);   // current tile landed; next still in flight
#endif
    __syncthreads();
    compute(buf);
    __syncthreads();
    buf ^= 1;
  }
#if USE_ASYNC_LDS
  __builtin_amdgcn_s_wait_asynccnt(0);
#endif
  __syncthreads();
  compute(buf);

  // epilogue
  const int grp = lane >> 4, li = lane & 15;
  const long r0 = (long)blockIdx.x * BM + wr * (BM / 2);
  const long c0 = (long)blockIdx.y * BN + wc * (BN / 4);
#pragma unroll
  for (int mi = 0; mi < MI; ++mi)
#pragma unroll
    for (int ni = 0; ni < NI; ++ni)
#pragma unroll
      for (int i = 0; i < 8; ++i) {
        long r = r0 + mi * 16 + i + grp * 8;
        long c = c0 + ni * 16 + li;
        float v = acc[mi][ni][i];
        if constexpr (EPI == 0) {
          ((bf16_t*)Cout)[(long)z * strideC + r * ldc + c] = (bf16_t)v;
        } else if constexpr (EPI == 1) {
          v = fmaxf(v + bias[c], 0.f);
          ((bf16_t*)Cout)[(long)z * strideC + r * ldc + c] = (bf16_t)v;
        } else if constexpr (EPI == 2) {
          ((float*)Cout)[(long)z * strideC + r * ldc + c] =
              Res[(long)z * strideR + r * ldc + c] + v;
        } else {
          ((float*)Cout)[(long)z * strideC + r * ldc + c] =
              Res[(long)z * strideR + r * ldc + c] + v + bias[c];
        }
      }
}

// ---------------------------------------------------------------------------
// Flash attention with r_mat softmax bias.
// grid (S/128, B*H); 8 waves x 16 query rows; 64-key tiles; WMMA for QK^T and PV.
// Output written permuted: feature index = d*H + h  (torch stack(dim=-1)).
// ---------------------------------------------------------------------------
__global__ __launch_bounds__(256) void attn_kernel(
    const bf16_t* __restrict__ Q, const bf16_t* __restrict__ Kb,
    const bf16_t* __restrict__ V, const float* __restrict__ r_mat,
    const float* __restrict__ attn_mask, const float* __restrict__ pad_mask,
    const float* __restrict__ rmx_g, const float* __restrict__ rsm_g,
    bf16_t* __restrict__ Aout) {
  const int z = blockIdx.y;                 // b*H + h
  const int b = z / H_, h = z % H_;
  const int row0 = blockIdx.x * 128;
  const int tid = threadIdx.x, lane = tid & 31, w = tid >> 5;
  const int grp = lane >> 4, li = lane & 15;

  __shared__ alignas(128) bf16_t sQ[128 * 80];
  __shared__ alignas(128) bf16_t sKT[64 * 80];   // transposed: [d][t]
  __shared__ alignas(128) bf16_t sV[64 * 80];    // natural:    [t][d]
  __shared__ alignas(128) bf16_t sP[8 * 16 * 80];

  const bf16_t* Qz = Q  + (long)z * S_ * Dh_;
  const bf16_t* Kz = Kb + (long)z * S_ * Dh_;
  const bf16_t* Vz = V  + (long)z * S_ * Dh_;

  // stage Q tile (128 x 64), async when available
#pragma unroll
  for (int it = 0; it < 4; ++it) {
    int c = tid + it * 256;
    int r = c >> 3, seg = c & 7;
    const void* gp = &Qz[(long)(row0 + r) * Dh_ + seg * 8];
    void* lp = &sQ[r * 80 + seg * 8];
#if USE_ASYNC_LDS
    async_copy16(gp, lp);
#else
    *(v8bf*)lp = *(const v8bf*)gp;
#endif
  }

  float m_run[8], l_run[8], rmx[8], rsm[8];
  v8f o_acc[4];
#pragma unroll
  for (int i = 0; i < 8; ++i) { m_run[i] = -3e38f; l_run[i] = 0.f; }
#pragma unroll
  for (int di = 0; di < 4; ++di) o_acc[di] = (v8f)0.f;
#pragma unroll
  for (int i = 0; i < 8; ++i) {
    int qr = row0 + w * 16 + i + grp * 8;
    rmx[i] = rmx_g[b * S_ + qr];
    rsm[i] = rsm_g[b * S_ + qr];
  }

  bf16_t* sPw = sP + w * 16 * 80;

  for (int kt = 0; kt < S_ / 64; ++kt) {
    const int t0 = kt * 64;
    __syncthreads();     // protect sKT/sV (and sP) from previous iteration
    // stage K transposed (scatter: scalar ds writes)
#pragma unroll
    for (int it = 0; it < 2; ++it) {
      int c = tid + it * 256;
      int t = c >> 3, seg = c & 7;
      v8bf kv = *(const v8bf*)&Kz[(long)(t0 + t) * Dh_ + seg * 8];
#pragma unroll
      for (int j = 0; j < 8; ++j) sKT[(seg * 8 + j) * 80 + t] = kv[j];
    }
    // stage V natural (async when available)
#pragma unroll
    for (int it = 0; it < 2; ++it) {
      int c = tid + it * 256;
      int t = c >> 3, seg = c & 7;
      const void* gp = &Vz[(long)(t0 + t) * Dh_ + seg * 8];
      void* lp = &sV[t * 80 + seg * 8];
#if USE_ASYNC_LDS
      async_copy16(gp, lp);
#else
      *(v8bf*)lp = *(const v8bf*)gp;
#endif
    }
#if USE_ASYNC_LDS
    __builtin_amdgcn_s_wait_asynccnt(0);   // covers sQ (first iter) and sV
#endif
    __syncthreads();

    // S = Q K^T for this wave's 16 rows x 64 keys
    v8f sacc[4];
#pragma unroll
    for (int ni = 0; ni < 4; ++ni) sacc[ni] = (v8f)0.f;
#pragma unroll
    for (int kc = 0; kc < 2; ++kc) {
      v16bf aq = frag_a(sQ, 80, w * 16, kc, lane);
#pragma unroll
      for (int ni = 0; ni < 4; ++ni) {
        v16bf bk = frag_b(sKT, 80, ni * 16, kc, lane);
        sacc[ni] = wmma_bf16(aq, bk, sacc[ni]);
      }
    }

    // scale, add r_mat softmax bias, mask, track tile row max
    float tmax[8];
#pragma unroll
    for (int i = 0; i < 8; ++i) tmax[i] = -3e38f;
#pragma unroll
    for (int i = 0; i < 8; ++i) {
      int qr = row0 + w * 16 + i + grp * 8;
      const float* rr = r_mat    + ((long)b * S_ + qr) * S_;
      const float* pr = pad_mask + ((long)b * S_ + qr) * S_;
      const float* ar = attn_mask + (long)qr * S_;
#pragma unroll
      for (int ni = 0; ni < 4; ++ni) {
        int t = t0 + ni * 16 + li;
        float s = sacc[ni][i] * SCALE_INV;
        bool msk = (pr[t] == 0.f) || (ar[t] == 0.f);
        float rv = msk ? NEGINF : rr[t];
        s += __expf(rv - rmx[i]) / rsm[i];
        if (msk) s = NEGINF;
        sacc[ni][i] = s;
        tmax[i] = fmaxf(tmax[i], s);
      }
    }
#pragma unroll
    for (int off = 8; off >= 1; off >>= 1)
#pragma unroll
      for (int i = 0; i < 8; ++i)
        tmax[i] = fmaxf(tmax[i], __shfl_xor(tmax[i], off, 32));

    float alpha[8], lsum[8];
#pragma unroll
    for (int i = 0; i < 8; ++i) {
      float mn = fmaxf(m_run[i], tmax[i]);
      alpha[i] = __expf(m_run[i] - mn);
      m_run[i] = mn;
      lsum[i] = 0.f;
    }
    // p = exp(s - m); write bf16 P into this wave's LDS slice
#pragma unroll
    for (int i = 0; i < 8; ++i)
#pragma unroll
      for (int ni = 0; ni < 4; ++ni) {
        float p = __expf(sacc[ni][i] - m_run[i]);
        lsum[i] += p;
        sPw[(i + grp * 8) * 80 + ni * 16 + li] = (bf16_t)p;
      }
#pragma unroll
    for (int off = 8; off >= 1; off >>= 1)
#pragma unroll
      for (int i = 0; i < 8; ++i) lsum[i] += __shfl_xor(lsum[i], off, 32);
#pragma unroll
    for (int i = 0; i < 8; ++i) l_run[i] = l_run[i] * alpha[i] + lsum[i];
#pragma unroll
    for (int di = 0; di < 4; ++di)
#pragma unroll
      for (int i = 0; i < 8; ++i) o_acc[di][i] *= alpha[i];

    // O += P V
#pragma unroll
    for (int kc = 0; kc < 2; ++kc) {
      v16bf ap = frag_a(sPw, 80, 0, kc, lane);
#pragma unroll
      for (int di = 0; di < 4; ++di) {
        v16bf bv = frag_b(sV, 80, di * 16, kc, lane);
        o_acc[di] = wmma_bf16(ap, bv, o_acc[di]);
      }
    }
  }

  // normalize and store permuted: feature = d*H + h
#pragma unroll
  for (int i = 0; i < 8; ++i) {
    int qr = row0 + w * 16 + i + grp * 8;
    float inv = 1.f / l_run[i];
#pragma unroll
    for (int di = 0; di < 4; ++di) {
      int d = di * 16 + li;
      Aout[((long)b * S_ + qr) * F_ + d * H_ + h] = (bf16_t)(o_acc[di][i] * inv);
    }
  }
}

// ---------------------------------------------------------------------------
// launcher
// ---------------------------------------------------------------------------
extern "C" void kernel_launch(void* const* d_in, const int* in_sizes, int n_in,
                              void* d_out, int out_size, void* d_ws, size_t ws_size,
                              hipStream_t stream) {
  const float* x         = (const float*)d_in[0];
  const float* r_mat     = (const float*)d_in[1];
  const float* attn_mask = (const float*)d_in[2];
  const float* pad_mask  = (const float*)d_in[3];
  const float* Wq        = (const float*)d_in[4];
  const float* Wk        = (const float*)d_in[5];
  const float* Wv        = (const float*)d_in[6];
  const float* Wo        = (const float*)d_in[7];
  const float* w1        = (const float*)d_in[8];
  const float* b1        = (const float*)d_in[9];
  const float* w2        = (const float*)d_in[10];
  const float* b2        = (const float*)d_in[11];
  const float* g1        = (const float*)d_in[12];
  const float* be1       = (const float*)d_in[13];
  const float* g2        = (const float*)d_in[14];
  const float* be2       = (const float*)d_in[15];
  float* out = (float*)d_out;

  char* ws = (char*)d_ws;
  size_t off = 0;
  auto carve = [&](size_t bytes) -> void* {
    void* p = ws + off;
    off += (bytes + 255) & ~(size_t)255;
    return p;
  };
  const long NW  = (long)H_ * F_ * Dh_;     // per-head weight elems (=1M)
  const long NSF = (long)B_ * S_ * F_;      // activation elems (=4M)
  bf16_t* wq_bf = (bf16_t*)carve(NW * 2);
  bf16_t* wk_bf = (bf16_t*)carve(NW * 2);
  bf16_t* wv_bf = (bf16_t*)carve(NW * 2);
  bf16_t* wo_bf = (bf16_t*)carve((long)F_ * F_ * 2);
  bf16_t* w1_bf = (bf16_t*)carve((long)F_ * EXPF * 2);
  bf16_t* w2_bf = (bf16_t*)carve((long)EXPF * F_ * 2);
  bf16_t* hbuf  = (bf16_t*)carve(NSF * 2);
  bf16_t* qbuf  = (bf16_t*)carve(NSF * 2);  // q,k,v,a contiguous: reused as FFN mid
  bf16_t* kbuf  = (bf16_t*)carve(NSF * 2);
  bf16_t* vbuf  = (bf16_t*)carve(NSF * 2);
  bf16_t* abuf  = (bf16_t*)carve(NSF * 2);
  float*  x1    = (float*)carve(NSF * 4);
  bf16_t* h2buf = (bf16_t*)carve(NSF * 2);
  float*  rmx   = (float*)carve((long)B_ * S_ * 4);
  float*  rsm   = (float*)carve((long)B_ * S_ * 4);
  bf16_t* mid   = qbuf;                      // 32MB alias over q/k/v/a (dead by FFN)
  (void)in_sizes; (void)n_in; (void)out_size; (void)ws_size;

  // 1) weights -> bf16
  cvt_bf16_kernel<<<NW / 1024, 256, 0, stream>>>(Wq, wq_bf, NW);
  cvt_bf16_kernel<<<NW / 1024, 256, 0, stream>>>(Wk, wk_bf, NW);
  cvt_bf16_kernel<<<NW / 1024, 256, 0, stream>>>(Wv, wv_bf, NW);
  cvt_bf16_kernel<<<(long)F_ * F_ / 1024, 256, 0, stream>>>(Wo, wo_bf, (long)F_ * F_);
  cvt_bf16_kernel<<<(long)F_ * EXPF / 1024, 256, 0, stream>>>(w1, w1_bf, (long)F_ * EXPF);
  cvt_bf16_kernel<<<(long)EXPF * F_ / 1024, 256, 0, stream>>>(w2, w2_bf, (long)EXPF * F_);

  // 2) LN1: x -> h (bf16)
  ln_kernel<<<B_ * S_, 256, 0, stream>>>(x, g1, be1, hbuf);

  // 3) Q/K/V projections: per (b,h) GEMM, M=S, N=Dh, K=F
  {
    dim3 g(S_ / 128, 1, B_ * H_);
    gemm_bf16_kernel<128, 64, 0><<<g, 256, 0, stream>>>(
        hbuf, F_, (long)S_ * F_, H_, wq_bf, Dh_, (long)F_ * Dh_, H_,
        qbuf, Dh_, (long)S_ * Dh_, nullptr, 0, nullptr, F_);
    gemm_bf16_kernel<128, 64, 0><<<g, 256, 0, stream>>>(
        hbuf, F_, (long)S_ * F_, H_, wk_bf, Dh_, (long)F_ * Dh_, H_,
        kbuf, Dh_, (long)S_ * Dh_, nullptr, 0, nullptr, F_);
    gemm_bf16_kernel<128, 64, 0><<<g, 256, 0, stream>>>(
        hbuf, F_, (long)S_ * F_, H_, wv_bf, Dh_, (long)F_ * Dh_, H_,
        vbuf, Dh_, (long)S_ * Dh_, nullptr, 0, nullptr, F_);
  }

  // 4) r_mat row stats
  rmstat_kernel<<<B_ * S_, 256, 0, stream>>>(r_mat, attn_mask, pad_mask, rmx, rsm);

  // 5) flash attention -> abuf (bf16, permuted d*H+h)
  attn_kernel<<<dim3(S_ / 128, B_ * H_), 256, 0, stream>>>(
      qbuf, kbuf, vbuf, r_mat, attn_mask, pad_mask, rmx, rsm, abuf);

  // 6) x1 = x + a @ Wo  (M = B*S flattened)
  gemm_bf16_kernel<128, 128, 2><<<dim3(B_ * S_ / 128, F_ / 128, 1), 256, 0, stream>>>(
      abuf, F_, 0, 1, wo_bf, F_, 0, 1, x1, F_, 0, x, 0, nullptr, F_);

  // 7) LN2: x1 -> h2 (bf16)
  ln_kernel<<<B_ * S_, 256, 0, stream>>>(x1, g2, be2, h2buf);

  // 8) mid = relu(h2 @ w1 + b1)  (bf16)
  gemm_bf16_kernel<128, 128, 1><<<dim3(B_ * S_ / 128, EXPF / 128, 1), 256, 0, stream>>>(
      h2buf, F_, 0, 1, w1_bf, EXPF, 0, 1, mid, EXPF, 0, nullptr, 0, b1, F_);

  // 9) out = x1 + mid @ w2 + b2  (fp32)
  gemm_bf16_kernel<128, 128, 3><<<dim3(B_ * S_ / 128, F_ / 128, 1), 256, 0, stream>>>(
      mid, EXPF, 0, 1, w2_bf, F_, 0, 1, out, F_, 0, x1, 0, b2, EXPF);
}
